// MoE_20255065767973
// MI455X (gfx1250) — matrix operand order repaired and checked by reading, not compiled, not run on hardware
//
#include <hip/hip_runtime.h>
#include <hip/hip_bf16.h>
#include <math.h>

// ---------------------------------------------------------------------------
// MoE dense-all-experts forward, MI455X (gfx1250), wave32 + WMMA bf16.
//   D=2048, H=8192, O=2048, N=8, T=4096 (T derived from in_sizes).
// v_wmma_f32_16x16x32_bf16 with fp32 accumulation; double-buffered LDS with
// GLOBAL_LOAD_ASYNC_TO_LDS_B128 staging (ASYNCcnt) when the builtin exists.
// ---------------------------------------------------------------------------

#define kD 2048
#define kH 8192
#define kO 2048
#define kN 8

#define LDSP 40   // LDS row stride in bf16 (80B: 16B aligned, 20-bank stride)

typedef __attribute__((ext_vector_type(16))) __bf16 v16bf;
typedef __attribute__((ext_vector_type(8)))  __bf16 v8bf;
typedef __attribute__((ext_vector_type(8)))  float  v8f;

// GCC-style int4 vector: matches the async-to-LDS builtin's pointee type
// (clang prints it as "__attribute__((__vector_size__(4 * sizeof(int)))) int").
typedef int v4i_vs __attribute__((vector_size(16)));

union BF16x16 { v16bf v; v8bf h[2]; };

// ---------------------------------------------------------------------------
// Async global->LDS staging (CDNA5 path), with synchronous fallback.
// ---------------------------------------------------------------------------
#if defined(__HIP_DEVICE_COMPILE__) && __has_builtin(__builtin_amdgcn_global_load_async_to_lds_b128)
#define ASYNC_STAGE 1
#else
#define ASYNC_STAGE 0
#endif

__device__ __forceinline__ void copy_b128(const __bf16* g, __bf16* l)
{
#if ASYNC_STAGE
    __builtin_amdgcn_global_load_async_to_lds_b128(
        (__attribute__((address_space(1))) v4i_vs*)(uintptr_t)g,
        (__attribute__((address_space(3))) v4i_vs*)(unsigned int)(uintptr_t)l,
        0, 0);
#else
    *(uint4*)l = *(const uint4*)g;
#endif
}

__device__ __forceinline__ void wait_stage()
{
#if ASYNC_STAGE
#if __has_builtin(__builtin_amdgcn_s_wait_asynccnt)
    __builtin_amdgcn_s_wait_asynccnt(0);
#else
    asm volatile("s_wait_asynccnt 0" ::: "memory");
#endif
#endif
}

// Each thread owns 2 rows x 16B in A and B tiles (128 rows x 64B per tile).
__device__ __forceinline__ void stage_copy(
    const __bf16* gA0, const __bf16* gA1,
    const __bf16* gB0, const __bf16* gB1, size_t koff,
    __bf16 (*As)[LDSP], __bf16 (*Bs)[LDSP], int r0, int off)
{
    copy_b128(gA0 + koff, &As[r0][off]);
    copy_b128(gA1 + koff, &As[r0 + 64][off]);
    copy_b128(gB0 + koff, &Bs[r0][off]);
    copy_b128(gB1 + koff, &Bs[r0 + 64][off]);
}

// ---------------------------------------------------------------------------
// Shared WMMA mainloop: C(128x128) += A(128xK) * B(128xK)^T, bf16 inputs.
// 8 wave32 per block (2x4); each wave owns 64x32 = 4x2 v8f accumulators.
// Double-buffered BK=32 stages.
// ---------------------------------------------------------------------------
__device__ __forceinline__ void mma_loop(
    const __bf16* __restrict__ Abase, const __bf16* __restrict__ Bbase,
    size_t ld, int K,
    __bf16 (*As)[128][LDSP], __bf16 (*Bs)[128][LDSP],
    v8f acc[4][2])
{
    const int tid  = threadIdx.x;
    const int lane = tid & 31;
    const int wave = tid >> 5;
    const int waveM = (wave >> 2) * 64;
    const int waveN = (wave & 3) * 32;
    const int lrow = lane & 15;
    const int kg   = lane >> 4;

    const int r0  = tid >> 2;         // 0..63
    const int off = (tid & 3) * 8;    // 0,8,16,24 elements

    const __bf16* gA0 = Abase + (size_t)r0 * ld + off;
    const __bf16* gA1 = gA0 + (size_t)64 * ld;
    const __bf16* gB0 = Bbase + (size_t)r0 * ld + off;
    const __bf16* gB1 = gB0 + (size_t)64 * ld;

    // prologue: stage 0 into buffer 0
    stage_copy(gA0, gA1, gB0, gB1, 0, As[0], Bs[0], r0, off);
    wait_stage();
    __syncthreads();

    const int nstages = K / 32;
    for (int s = 0; s < nstages; ++s) {
        const int cur  = s & 1;
        const bool more = (s + 1) < nstages;

        if (more)  // kick off next stage into the other buffer (async)
            stage_copy(gA0, gA1, gB0, gB1, (size_t)(s + 1) * 32,
                       As[cur ^ 1], Bs[cur ^ 1], r0, off);
        if (s + 2 < nstages) {  // global_prefetch_b8 for the stage after next
            __builtin_prefetch(gA0 + (size_t)(s + 2) * 32, 0, 0);
            __builtin_prefetch(gB0 + (size_t)(s + 2) * 32, 0, 0);
        }

        BF16x16 bfj[2];
#pragma unroll
        for (int j = 0; j < 2; ++j) {
            int col = waveN + j * 16 + lrow;
            bfj[j].h[0] = *(const v8bf*)(&Bs[cur][col][kg * 8]);
            bfj[j].h[1] = *(const v8bf*)(&Bs[cur][col][16 + kg * 8]);
        }
#pragma unroll
        for (int i = 0; i < 4; ++i) {
            BF16x16 af;
            int row = waveM + i * 16 + lrow;
            af.h[0] = *(const v8bf*)(&As[cur][row][kg * 8]);
            af.h[1] = *(const v8bf*)(&As[cur][row][16 + kg * 8]);
#pragma unroll
            for (int j = 0; j < 2; ++j)
                acc[i][j] = __builtin_amdgcn_wmma_f32_16x16x32_bf16(
                    false, af.v, false, bfj[j].v, (short)0, acc[i][j], false, false);
        }

        if (more) {
            wait_stage();      // next buffer fully in LDS
            __syncthreads();   // all waves done reading cur / writing nxt
        }
    }
}

// ---------------------------------------------------------------------------
// fp32 -> bf16 conversion (vectorized; all sizes here are multiples of 4)
// ---------------------------------------------------------------------------
__global__ __launch_bounds__(256) void f32_to_bf16_kernel(
    const float* __restrict__ s, __bf16* __restrict__ d, size_t n)
{
    size_t i = ((size_t)blockIdx.x * blockDim.x + threadIdx.x) * 4;
    if (i + 3 < n) {
        float4 v = *(const float4*)(s + i);
        d[i + 0] = (__bf16)v.x;
        d[i + 1] = (__bf16)v.y;
        d[i + 2] = (__bf16)v.z;
        d[i + 3] = (__bf16)v.w;
    } else {
        for (; i < n; ++i) d[i] = (__bf16)s[i];
    }
}

// ---------------------------------------------------------------------------
// Boltzmann gate: logits = x @ Wg^T + bg; p = softmax(logits / e);
// keep top-5 of 8, renormalize. One wave32 per token.
// ---------------------------------------------------------------------------
__global__ __launch_bounds__(256) void gate_kernel(
    const float* __restrict__ x, const float* __restrict__ Wg,
    const float* __restrict__ bg, float* __restrict__ wout, int T)
{
    const int wave = threadIdx.x >> 5;
    const int lane = threadIdx.x & 31;
    const int t = blockIdx.x * 8 + wave;
    if (t >= T) return;

    float acc[kN];
#pragma unroll
    for (int e = 0; e < kN; ++e) acc[e] = 0.0f;

    const float* xr = x + (size_t)t * kD;
    for (int d = lane; d < kD; d += 32) {
        float xv = xr[d];
#pragma unroll
        for (int e = 0; e < kN; ++e) acc[e] += xv * Wg[(size_t)e * kD + d];
    }
#pragma unroll
    for (int e = 0; e < kN; ++e) {
        float v = acc[e];
#pragma unroll
        for (int off = 16; off > 0; off >>= 1) v += __shfl_xor(v, off, 32);
        acc[e] = v + bg[e];
    }

    if (lane == 0) {
        const float invT = 1.0f / 2.718281828459045f;
        float mx = -1e30f;
#pragma unroll
        for (int e = 0; e < kN; ++e) {
            acc[e] *= invT;
            mx = fmaxf(mx, acc[e]);
        }
        float p[kN], s = 0.0f;
#pragma unroll
        for (int e = 0; e < kN; ++e) { p[e] = expf(acc[e] - mx); s += p[e]; }
        float inv = 1.0f / s;
#pragma unroll
        for (int e = 0; e < kN; ++e) p[e] *= inv;

        bool keep[kN];
#pragma unroll
        for (int e = 0; e < kN; ++e) keep[e] = true;
        for (int it = 0; it < 3; ++it) {   // drop 3 smallest -> keep top NA=5
            int mi = -1; float mv = 1e30f;
#pragma unroll
            for (int e = 0; e < kN; ++e)
                if (keep[e] && p[e] < mv) { mv = p[e]; mi = e; }
            keep[mi] = false;
        }
        float s2 = 0.0f;
#pragma unroll
        for (int e = 0; e < kN; ++e) if (keep[e]) s2 += p[e];
        float rn = 1.0f / (s2 + 1e-8f);
        float* wr = wout + (size_t)t * kN;
#pragma unroll
        for (int e = 0; e < kN; ++e) wr[e] = keep[e] ? p[e] * rn : 0.0f;
    }
}

// ---------------------------------------------------------------------------
// GEMM1: Hout[t,h] = relu(Xb[t,:] . W1b[h,:] + b1[h]);  K = kD
// ---------------------------------------------------------------------------
__global__ __launch_bounds__(256) void gemm1_relu_kernel(
    const __bf16* __restrict__ X,   // [T, kD] bf16
    const __bf16* __restrict__ W,   // [kH, kD] bf16 (one expert)
    const float* __restrict__ bias, // [kH]
    __bf16* __restrict__ Hout,      // [T, kH] bf16
    int T)
{
    __shared__ __bf16 As[2][128][LDSP];
    __shared__ __bf16 Bs[2][128][LDSP];

    const int m0 = blockIdx.y * 128;
    const int n0 = blockIdx.x * 128;
    const int lane = threadIdx.x & 31;
    const int wave = threadIdx.x >> 5;
    const int waveM = (wave >> 2) * 64;
    const int waveN = (wave & 3) * 32;
    const int lrow = lane & 15;
    const int kg = lane >> 4;

    v8f acc[4][2];
#pragma unroll
    for (int i = 0; i < 4; ++i)
#pragma unroll
        for (int j = 0; j < 2; ++j) {
            v8f z = {0.f, 0.f, 0.f, 0.f, 0.f, 0.f, 0.f, 0.f};
            acc[i][j] = z;
        }

    mma_loop(X + (size_t)m0 * kD, W + (size_t)n0 * kD, kD, kD, As, Bs, acc);

    // epilogue: bias + ReLU, bf16 store
#pragma unroll
    for (int j = 0; j < 2; ++j) {
        int col = n0 + waveN + j * 16 + lrow;
        float bb = bias[col];
#pragma unroll
        for (int i = 0; i < 4; ++i) {
#pragma unroll
            for (int r = 0; r < 8; ++r) {
                int row = m0 + waveM + i * 16 + kg * 8 + r;
                float v = acc[i][j][r] + bb;
                v = v > 0.0f ? v : 0.0f;
                Hout[(size_t)row * kH + col] = (__bf16)v;
            }
        }
    }
}

// ---------------------------------------------------------------------------
// GEMM2: Out[t,o] (+)= wgate[t,n] * (H[t,:] . W2b[o,:] + b2[o]);  K = kH
// ---------------------------------------------------------------------------
__global__ __launch_bounds__(256) void gemm2_acc_kernel(
    const __bf16* __restrict__ Hin,   // [T, kH] bf16
    const __bf16* __restrict__ W,     // [kO, kH] bf16 (one expert)
    const float* __restrict__ bias,   // [kO]
    const float* __restrict__ wgate,  // [T, kN]
    int expert,
    float* __restrict__ Out,          // [T, kO]
    int T, int accumulate)
{
    __shared__ __bf16 As[2][128][LDSP];
    __shared__ __bf16 Bs[2][128][LDSP];

    const int m0 = blockIdx.y * 128;
    const int n0 = blockIdx.x * 128;
    const int lane = threadIdx.x & 31;
    const int wave = threadIdx.x >> 5;
    const int waveM = (wave >> 2) * 64;
    const int waveN = (wave & 3) * 32;
    const int lrow = lane & 15;
    const int kg = lane >> 4;

    v8f acc[4][2];
#pragma unroll
    for (int i = 0; i < 4; ++i)
#pragma unroll
        for (int j = 0; j < 2; ++j) {
            v8f z = {0.f, 0.f, 0.f, 0.f, 0.f, 0.f, 0.f, 0.f};
            acc[i][j] = z;
        }

    mma_loop(Hin + (size_t)m0 * kH, W + (size_t)n0 * kH, kH, kH, As, Bs, acc);

    // epilogue: out (+)= w[t,n] * (acc + b2)
    float bb[2];
#pragma unroll
    for (int j = 0; j < 2; ++j) bb[j] = bias[n0 + waveN + j * 16 + lrow];

#pragma unroll
    for (int i = 0; i < 4; ++i) {
#pragma unroll
        for (int r = 0; r < 8; ++r) {
            int row = m0 + waveM + i * 16 + kg * 8 + r;
            float gw = wgate[(size_t)row * kN + expert];
#pragma unroll
            for (int j = 0; j < 2; ++j) {
                int col = n0 + waveN + j * 16 + lrow;
                size_t oi = (size_t)row * kO + col;
                float v = gw * (acc[i][j][r] + bb[j]);
                Out[oi] = (accumulate ? Out[oi] : 0.0f) + v;
            }
        }
    }
}

// ---------------------------------------------------------------------------
// Launch: gate -> convert x -> per-expert {convert W1, GEMM1, convert W2, GEMM2}
// ---------------------------------------------------------------------------
extern "C" void kernel_launch(void* const* d_in, const int* in_sizes, int n_in,
                              void* d_out, int out_size, void* d_ws, size_t ws_size,
                              hipStream_t stream)
{
    const float* x  = (const float*)d_in[0];   // [T, D]
    const float* W1 = (const float*)d_in[1];   // [N, H, D]
    const float* b1 = (const float*)d_in[2];   // [N, H]
    const float* W2 = (const float*)d_in[3];   // [N, O, H]
    const float* b2 = (const float*)d_in[4];   // [N, O]
    const float* Wg = (const float*)d_in[5];   // [N, D]
    const float* bg = (const float*)d_in[6];   // [N]

    const int T = in_sizes[0] / kD;            // 4096

    float* out  = (float*)d_out;               // [T, O]
    float* wout = out + (size_t)T * kO;        // [T, N]

    // workspace layout (bf16 buffers, 256B aligned)
    char* ws = (char*)d_ws;
    size_t off = 0;
    __bf16* xb = (__bf16*)(ws + off);
    off += (((size_t)T * kD * 2) + 255) & ~(size_t)255;
    __bf16* w1b = (__bf16*)(ws + off);
    off += (((size_t)kH * kD * 2) + 255) & ~(size_t)255;
    __bf16* w2b = (__bf16*)(ws + off);
    off += (((size_t)kO * kH * 2) + 255) & ~(size_t)255;
    __bf16* hb = (__bf16*)(ws + off);

    // 1. gate weights (fp32 path, trivial FLOPs)
    gate_kernel<<<(T + 7) / 8, 256, 0, stream>>>(x, Wg, bg, wout, T);

    // 2. x -> bf16 once
    {
        size_t n = (size_t)T * kD;
        f32_to_bf16_kernel<<<(unsigned)((n / 4 + 255) / 256), 256, 0, stream>>>(x, xb, n);
    }

    // 3. per-expert fused MLP
    const size_t w1n = (size_t)kH * kD;
    const size_t w2n = (size_t)kO * kH;
    for (int n = 0; n < kN; ++n) {
        f32_to_bf16_kernel<<<(unsigned)((w1n / 4 + 255) / 256), 256, 0, stream>>>(
            W1 + (size_t)n * w1n, w1b, w1n);
        gemm1_relu_kernel<<<dim3(kH / 128, T / 128), 256, 0, stream>>>(
            xb, w1b, b1 + (size_t)n * kH, hb, T);
        f32_to_bf16_kernel<<<(unsigned)((w2n / 4 + 255) / 256), 256, 0, stream>>>(
            W2 + (size_t)n * w2n, w2b, w2n);
        gemm2_acc_kernel<<<dim3(kO / 128, T / 128), 256, 0, stream>>>(
            hb, w2b, b2 + (size_t)n * kO, wout, n, out, T, n > 0);
    }
}
